// Faster_RCNN_18399639896855
// MI455X (gfx1250) — compile-verified, matrix operand order checked by reading, not run on hardware
//
#include <hip/hip_runtime.h>
#include <math.h>

// ---------------- types ----------------
typedef _Float16 v16h __attribute__((ext_vector_type(16)));
typedef _Float16 v8h  __attribute__((ext_vector_type(8)));
typedef float    v8f  __attribute__((ext_vector_type(8)));

#define NEGC (-1000000000.0f)

// ---------------- f32 -> f16 convert ----------------
__global__ void cvt_f32_f16(const float* __restrict__ in, _Float16* __restrict__ out, long n) {
    long id = (long)blockIdx.x * blockDim.x + threadIdx.x;
    if (id < n) out[id] = (_Float16)in[id];
}

// ---------------- pack B f32[KxN] -> f16 WMMA-fragment layout ----------------
// Bp[((nt*(K/32)+kt)*32 + lane)*16 + i] = B[(kt*32 + kfrag(lane,i))*N + nt*16 + lane%16]
// kfrag: lanes 0-15 hold K {0..7,16..23}, lanes 16-31 hold K {8..15,24..31}
__global__ void pack_b_f16(const float* __restrict__ B, _Float16* __restrict__ Bp,
                           int K, int N) {
    long total = (long)K * N;
    long id = (long)blockIdx.x * blockDim.x + threadIdx.x;
    if (id >= total) return;
    int i = (int)(id & 15);
    long t = id >> 4;
    int lane = (int)(t & 31); t >>= 5;
    int kt = (int)(t % (K >> 5));
    int nt = (int)(t / (K >> 5));
    int hi8 = (lane >> 4) * 8;
    int k = (i < 8) ? (i + hi8) : (i + 8 + hi8);
    int gk = (kt << 5) + k;
    int gn = (nt << 4) + (lane & 15);
    Bp[id] = (_Float16)B[(size_t)gk * N + gn];
}

// ---------------- 3x3 SAME im2col, f32 in -> f16 out ----------------
__global__ void im2col3x3(const float* __restrict__ in, _Float16* __restrict__ out,
                          int H, int W, int C) {
    long total = (long)H * W * 9 * C;
    long id = (long)blockIdx.x * blockDim.x + threadIdx.x;
    if (id >= total) return;
    int c = (int)(id % C);
    long t = id / C;
    int k = (int)(t % 9); t /= 9;
    int x = (int)(t % W);
    int y = (int)(t / W);
    int ky = k / 3, kx = k % 3;
    int sy = y + ky - 1, sx = x + kx - 1;
    float v = 0.0f;
    if (sy >= 0 && sy < H && sx >= 0 && sx < W)
        v = in[((long)sy * W + sx) * C + c];
    out[id] = (_Float16)v;
}

// ---------------- WMMA GEMM: C[MxN] = A_f16[MxK] @ Bpacked_f16 (+bias,bn,relu) -----
// Wave tile 16(M) x 64(N): one A fragment drives 4 WMMAs per 32-wide k-step.
// Block = 256 threads = 8 waves -> 128 rows; grid = (N/64, ceil(M/128)).
// No LDS, no barriers, no converts in the hot loop.
__launch_bounds__(256)
__global__ void wmma_gemm4(const _Float16* __restrict__ A, const _Float16* __restrict__ Bp,
                           float* __restrict__ C,
                           const float* __restrict__ bias,
                           const float* __restrict__ gamma, const float* __restrict__ beta,
                           const float* __restrict__ mean,  const float* __restrict__ var,
                           int M, int N, int K, int flags /*1=relu 2=bn*/) {
    const int lane = threadIdx.x & 31;
    const int wave = threadIdx.x >> 5;
    const int nb = blockIdx.x * 64;
    const int mb = blockIdx.y * 128 + wave * 16;
    const int l  = lane & 15;
    const bool hi = lane >= 16;

    int row = mb + l;
    if (row >= M) row = M - 1;            // clamp: EXEC must stay all-1s for WMMA
    const _Float16* Ap = A + (size_t)row * K + (hi ? 8 : 0);

    const int ktot = K >> 5;
    const size_t tstride = (size_t)ktot * 512;      // f16 elems per 16-col tile
    const _Float16* B0 = Bp + (size_t)((nb >> 4) + 0) * tstride + lane * 16;
    const _Float16* B1 = B0 + tstride;
    const _Float16* B2 = B1 + tstride;
    const _Float16* B3 = B2 + tstride;

    v8f acc0 = {}, acc1 = {}, acc2 = {}, acc3 = {};
    for (int kt = 0; kt < ktot; ++kt) {
        const _Float16* Ar = Ap + (kt << 5);
        v8h a0 = *(const v8h*)(Ar);
        v8h a1 = *(const v8h*)(Ar + 16);
        if (kt + 2 < ktot) __builtin_prefetch(Ar + 64, 0, 1);
        v16h a;
#pragma unroll
        for (int i = 0; i < 8; ++i) { a[i] = a0[i]; a[i + 8] = a1[i]; }

        size_t boff = (size_t)kt * 512;
        v16h b0 = *(const v16h*)(B0 + boff);
        v16h b1 = *(const v16h*)(B1 + boff);
        v16h b2 = *(const v16h*)(B2 + boff);
        v16h b3 = *(const v16h*)(B3 + boff);

        acc0 = __builtin_amdgcn_wmma_f32_16x16x32_f16(false, a, false, b0, (short)0, acc0, false, false);
        acc1 = __builtin_amdgcn_wmma_f32_16x16x32_f16(false, a, false, b1, (short)0, acc1, false, false);
        acc2 = __builtin_amdgcn_wmma_f32_16x16x32_f16(false, a, false, b2, (short)0, acc2, false, false);
        acc3 = __builtin_amdgcn_wmma_f32_16x16x32_f16(false, a, false, b3, (short)0, acc3, false, false);
    }

    v8f accs[4] = {acc0, acc1, acc2, acc3};
#pragma unroll
    for (int s = 0; s < 4; ++s) {
        int col = nb + s * 16 + l;
        if (col >= N) continue;
        float bv = bias ? bias[col] : 0.0f;
        float sc = 1.0f, sh = 0.0f;
        if (flags & 2) {
            sc = gamma[col] * rsqrtf(var[col] + 1e-3f);
            sh = beta[col] - mean[col] * sc;
        }
#pragma unroll
        for (int r = 0; r < 8; ++r) {
            int m = mb + r + (hi ? 8 : 0);
            if (m < M) {
                float v = accs[s][r] + bv;
                if (flags & 2) v = v * sc + sh;
                if (flags & 1) v = fmaxf(v, 0.0f);
                C[(size_t)m * N + col] = v;
            }
        }
    }
}

// ---------------- small dense: C = A f32 [MxK] @ W f32 [KxN] + b ----------------
__global__ void dense_bias(const float* __restrict__ A, const float* __restrict__ W,
                           const float* __restrict__ b, float* __restrict__ C,
                           int M, int N, int K) {
    long id = (long)blockIdx.x * blockDim.x + threadIdx.x;
    if (id >= (long)M * N) return;
    int col = (int)(id % N);
    long row = id / N;
    const float* a = A + row * K;
    float s = b[col];
    for (int k = 0; k < K; ++k) s += a[k] * W[(long)k * N + col];
    C[id] = s;
}

// ---------------- RPN 2-way softmax + score extraction ----------------
__global__ void rpn_softmax(const float* __restrict__ logits, float* __restrict__ probs,
                            float* __restrict__ scores, int n) {
    int i = blockIdx.x * blockDim.x + threadIdx.x;
    if (i >= n) return;
    float l0 = logits[2 * i], l1 = logits[2 * i + 1];
    float m = fmaxf(l0, l1);
    float e0 = expf(l0 - m), e1 = expf(l1 - m);
    float d = 1.0f / (e0 + e1);
    probs[2 * i]     = e0 * d;
    probs[2 * i + 1] = e1 * d;
    scores[i]        = e1 * d;
}

// ---------------- exact sorted top-k by rank (ties -> lower index) ----------------
__global__ void topk_select(const float* __restrict__ s, float* __restrict__ tv,
                            int* __restrict__ ti, int n, int k) {
    int i = blockIdx.x * blockDim.x + threadIdx.x;
    if (i >= n) return;
    float si = s[i];
    int rank = 0;
    for (int j = 0; j < n; ++j) {
        float sj = s[j];
        rank += (sj > si) || (sj == si && j < i);
        if (rank >= k) return;
    }
    tv[rank] = si;
    ti[rank] = i;
}

// ---------------- anchor generation + delta decode + clip/normalize ----------------
__global__ void decode_props(const int* __restrict__ ti, const float* __restrict__ tv,
                             const float* __restrict__ deltas, float* __restrict__ props,
                             float* __restrict__ nms_sc,
                             const int* __restrict__ ph, const int* __restrict__ pw,
                             int n, int fw) {
    int p = blockIdx.x * blockDim.x + threadIdx.x;
    if (p >= n) return;
    float H = (float)(*ph), W = (float)(*pw);
    int idx = ti[p];
    int a = idx % 9, cell = idx / 9;
    int x = cell % fw, y = cell / fw;
    const float SC[3] = {64.0f, 128.0f, 256.0f};
    const float RT[3] = {0.5f, 1.0f, 2.0f};
    float sq = sqrtf(RT[a % 3]);
    float ah = SC[a / 3] * sq, aw = SC[a / 3] / sq;
    float cy = ((float)y + 0.5f) * (H / (float)fw);
    float cx = ((float)x + 0.5f) * (W / (float)fw);
    float y1 = cy - 0.5f * ah, x1 = cx - 0.5f * aw;
    float h = ah, w = aw;
    float dy = deltas[idx * 4 + 0], dx = deltas[idx * 4 + 1];
    float dh = deltas[idx * 4 + 2], dw = deltas[idx * 4 + 3];
    float ncy = y1 + 0.5f * h + dy * h;
    float ncx = x1 + 0.5f * w + dx * w;
    float nh = h * expf(dh), nw = w * expf(dw);
    float b0 = fminf(fmaxf(ncy - 0.5f * nh, 0.0f), H) / H;
    float b1 = fminf(fmaxf(ncx - 0.5f * nw, 0.0f), W) / W;
    float b2 = fminf(fmaxf(ncy + 0.5f * nh, 0.0f), H) / H;
    float b3 = fminf(fmaxf(ncx + 0.5f * nw, 0.0f), W) / W;
    props[p * 4 + 0] = b0; props[p * 4 + 1] = b1;
    props[p * 4 + 2] = b2; props[p * 4 + 3] = b3;
    nms_sc[p] = tv[p];
}

// ---------------- sequential NMS (one workgroup, LDS argmax reductions) ----------------
__launch_bounds__(256)
__global__ void nms_kernel(const float* __restrict__ props, float* __restrict__ sc, int n,
                           int* __restrict__ sel, int* __restrict__ valid,
                           int iters, float thr) {
    __shared__ float bs[256];
    __shared__ int   bi[256];
    __shared__ float box[4];
    for (int it = 0; it < iters; ++it) {
        float best = -INFINITY; int bidx = 0x7fffffff;
        for (int j = threadIdx.x; j < n; j += 256) {
            float s = sc[j];
            if (s > best || (s == best && j < bidx)) { best = s; bidx = j; }
        }
        bs[threadIdx.x] = best; bi[threadIdx.x] = bidx;
        __syncthreads();
        if (threadIdx.x == 0) {
            float b = bs[0]; int i0 = bi[0];
            for (int t = 1; t < 256; ++t)
                if (bs[t] > b || (bs[t] == b && bi[t] < i0)) { b = bs[t]; i0 = bi[t]; }
            sel[it]   = i0;
            valid[it] = (b > NEGC * 0.5f) ? 1 : 0;
            box[0] = props[i0 * 4 + 0]; box[1] = props[i0 * 4 + 1];
            box[2] = props[i0 * 4 + 2]; box[3] = props[i0 * 4 + 3];
            sc[i0] = NEGC;
        }
        __syncthreads();
        float by1 = box[0], bx1 = box[1], by2 = box[2], bx2 = box[3];
        float ba = (by2 - by1) * (bx2 - bx1);
        for (int j = threadIdx.x; j < n; j += 256) {
            float jy1 = props[j * 4 + 0], jx1 = props[j * 4 + 1];
            float jy2 = props[j * 4 + 2], jx2 = props[j * 4 + 3];
            float y1 = fmaxf(by1, jy1), x1 = fmaxf(bx1, jx1);
            float y2 = fminf(by2, jy2), x2 = fminf(bx2, jx2);
            float inter = fmaxf(y2 - y1, 0.0f) * fmaxf(x2 - x1, 0.0f);
            float ja = (jy2 - jy1) * (jx2 - jx1);
            float iou = inter / (ba + ja - inter + 1e-8f);
            if (iou > thr) sc[j] = NEGC;
        }
        __syncthreads();
    }
}

// ---------------- gather selected proposals ----------------
__global__ void gather_props(const float* __restrict__ props, const int* __restrict__ sel,
                             const int* __restrict__ valid, float* __restrict__ out_prop,
                             float* __restrict__ ws_prop, int n) {
    int p = blockIdx.x * blockDim.x + threadIdx.x;
    if (p >= n) return;
    int i = sel[p];
    float f = valid[p] ? 1.0f : 0.0f;
    for (int j = 0; j < 4; ++j) {
        float v = props[i * 4 + j] * f;
        out_prop[p * 4 + j] = v;
        ws_prop[p * 4 + j]  = v;
    }
}

// ---------------- ROI-align (bilinear, 14x14), emits f16 A-matrix for FC1 ----------------
__global__ void roi_align_kernel(const float* __restrict__ feat, const float* __restrict__ boxes,
                                 _Float16* __restrict__ out, int Hf, int Wf, int C, int pool) {
    int c    = threadIdx.x;     // C == blockDim.x
    int cell = blockIdx.x;      // pool*pool
    int p    = blockIdx.y;
    int iy = cell / pool, ix = cell % pool;
    float y1 = boxes[p * 4 + 0], x1 = boxes[p * 4 + 1];
    float y2 = boxes[p * 4 + 2], x2 = boxes[p * 4 + 3];
    float ty = (float)iy / (float)(pool - 1);
    float tx = (float)ix / (float)(pool - 1);
    float ys = (y1 + ty * (y2 - y1)) * (float)(Hf - 1);
    float xs = (x1 + tx * (x2 - x1)) * (float)(Wf - 1);
    int y0 = (int)fminf(fmaxf(floorf(ys), 0.0f), (float)(Hf - 2));
    int x0 = (int)fminf(fmaxf(floorf(xs), 0.0f), (float)(Wf - 2));
    float wy = fminf(fmaxf(ys - (float)y0, 0.0f), 1.0f);
    float wx = fminf(fmaxf(xs - (float)x0, 0.0f), 1.0f);
    const float* f00 = feat + ((long)y0 * Wf + x0) * C + c;
    float v00 = f00[0], v01 = f00[C];
    float v10 = f00[(long)Wf * C], v11 = f00[(long)Wf * C + C];
    float top = v00 * (1.0f - wx) + v01 * wx;
    float bot = v10 * (1.0f - wx) + v11 * wx;
    float v   = top * (1.0f - wy) + bot * wy;
    out[((long)p * (pool * pool) + cell) * C + c] = (_Float16)v;
}

// ---------------- row softmax (class probs) ----------------
__global__ void softmax_rows(const float* __restrict__ in, float* __restrict__ out,
                             int M, int N) {
    int r = blockIdx.x * blockDim.x + threadIdx.x;
    if (r >= M) return;
    const float* a = in + (long)r * N;
    float* o = out + (long)r * N;
    float m = -INFINITY;
    for (int n = 0; n < N; ++n) m = fmaxf(m, a[n]);
    float s = 0.0f;
    for (int n = 0; n < N; ++n) s += expf(a[n] - m);
    float inv = 1.0f / s;
    for (int n = 0; n < N; ++n) o[n] = expf(a[n] - m) * inv;
}

// =======================================================================
extern "C" void kernel_launch(void* const* d_in, const int* in_sizes, int n_in,
                              void* d_out, int out_size, void* d_ws, size_t ws_size,
                              hipStream_t stream) {
    const float* feature = (const float*)d_in[0];
    const float* w_fpn1  = (const float*)d_in[1];
    const float* b_fpn1  = (const float*)d_in[2];
    const float* w_fpn2  = (const float*)d_in[3];
    const float* b_fpn2  = (const float*)d_in[4];
    const float* w_rpn   = (const float*)d_in[5];
    const float* b_rpn   = (const float*)d_in[6];
    const float* w_cls   = (const float*)d_in[7];
    const float* b_cls   = (const float*)d_in[8];
    const float* w_dlt   = (const float*)d_in[9];
    const float* b_dlt   = (const float*)d_in[10];
    const float* w_rc1   = (const float*)d_in[11];
    const float* b_rc1   = (const float*)d_in[12];
    const float* g1      = (const float*)d_in[13];
    const float* be1     = (const float*)d_in[14];
    const float* m1      = (const float*)d_in[15];
    const float* v1      = (const float*)d_in[16];
    const float* w_rc2   = (const float*)d_in[17];
    const float* b_rc2   = (const float*)d_in[18];
    const float* g2      = (const float*)d_in[19];
    const float* be2     = (const float*)d_in[20];
    const float* m2      = (const float*)d_in[21];
    const float* v2      = (const float*)d_in[22];
    const float* w_log   = (const float*)d_in[23];
    const float* b_log   = (const float*)d_in[24];
    const float* w_bfc   = (const float*)d_in[25];
    const float* b_bfc   = (const float*)d_in[26];
    const int*   image_h = (const int*)d_in[27];
    const int*   image_w = (const int*)d_in[28];

    const int FH = 64, FW = 64, NPIX = FH * FW;          // 4096
    const int CIN = 1024, CF = 256, CR = 512;
    const int NA = NPIX * 9;                              // 36864
    const int PRE = 6000, PC = 1000, POOL = 14;
    const int KROI = POOL * POOL * CF;                    // 50176
    const int NCLS = 81;

    // output offsets (flat f32, concatenated in return order)
    float* out = (float*)d_out;
    float* out_probs = out;                               // 73728
    float* out_delta = out + 73728;                       // 147456
    float* out_prop  = out + 73728 + 147456;              // 4000
    float* out_cls   = out_prop + 4000;                   // 81000
    float* out_box   = out_cls + 81000;                   // 324000

    // workspace bump allocator
    char* wsb = (char*)d_ws;
    size_t off = 0;
    auto alloc = [&](size_t bytes) -> void* {
        off = (off + 255) & ~(size_t)255;
        void* p = wsb + off;
        off += bytes;
        return p;
    };
    _Float16* feat16  = (_Float16*)alloc((size_t)NPIX * CIN * 2);
    float*    x1      = (float*)   alloc((size_t)NPIX * CF * 4);
    _Float16* col16   = (_Float16*)alloc((size_t)NPIX * 9 * CF * 2);
    float*    xbuf    = (float*)   alloc((size_t)NPIX * CF * 4);
    float*    shbuf   = (float*)   alloc((size_t)NPIX * CR * 4);
    float*    rpn_cl  = (float*)   alloc((size_t)NA * 2 * 4);
    float*    scores  = (float*)   alloc((size_t)NA * 4);
    float*    tv      = (float*)   alloc((size_t)PRE * 4);
    int*      ti      = (int*)     alloc((size_t)PRE * 4);
    float*    props   = (float*)   alloc((size_t)PRE * 4 * 4);
    float*    nsc     = (float*)   alloc((size_t)PRE * 4);
    int*      sel     = (int*)     alloc((size_t)PC * 4);
    int*      vld     = (int*)     alloc((size_t)PC * 4);
    float*    prop_ws = (float*)   alloc((size_t)PC * 4 * 4);
    _Float16* rois16  = (_Float16*)alloc((size_t)PC * KROI * 2);
    float*    h1      = (float*)   alloc((size_t)PC * 1024 * 4);
    _Float16* h1f16   = (_Float16*)alloc((size_t)PC * 1024 * 2);
    float*    h2      = (float*)   alloc((size_t)PC * 1024 * 4);
    float*    clsl    = (float*)   alloc((size_t)PC * NCLS * 4);
    _Float16* bpack   = (_Float16*)alloc((size_t)KROI * 1024 * 2);  // reused for all 5 GEMMs
    (void)ws_size; (void)n_in; (void)in_sizes; (void)out_size;

    // 1) feature -> f16
    {
        long n = (long)NPIX * CIN;
        cvt_f32_f16<<<(unsigned)((n + 255) / 256), 256, 0, stream>>>(feature, feat16, n);
    }
    // 2) fpn1 1x1: x1 = feat @ w_fpn1 + b
    {
        long n = (long)CIN * CF;
        pack_b_f16<<<(unsigned)((n + 255) / 256), 256, 0, stream>>>(w_fpn1, bpack, CIN, CF);
    }
    wmma_gemm4<<<dim3(CF / 64, NPIX / 128), 256, 0, stream>>>(
        feat16, bpack, x1, b_fpn1, nullptr, nullptr, nullptr, nullptr,
        NPIX, CF, CIN, 0);
    // 3) im2col(x1), 4) fpn2 3x3: xbuf
    {
        long n = (long)NPIX * 9 * CF;
        im2col3x3<<<(unsigned)((n + 255) / 256), 256, 0, stream>>>(x1, col16, FH, FW, CF);
        n = (long)(9 * CF) * CF;
        pack_b_f16<<<(unsigned)((n + 255) / 256), 256, 0, stream>>>(w_fpn2, bpack, 9 * CF, CF);
    }
    wmma_gemm4<<<dim3(CF / 64, NPIX / 128), 256, 0, stream>>>(
        col16, bpack, xbuf, b_fpn2, nullptr, nullptr, nullptr, nullptr,
        NPIX, CF, 9 * CF, 0);
    // 5) im2col(xbuf), 6) rpn 3x3 + relu: shbuf
    {
        long n = (long)NPIX * 9 * CF;
        im2col3x3<<<(unsigned)((n + 255) / 256), 256, 0, stream>>>(xbuf, col16, FH, FW, CF);
        n = (long)(9 * CF) * CR;
        pack_b_f16<<<(unsigned)((n + 255) / 256), 256, 0, stream>>>(w_rpn, bpack, 9 * CF, CR);
    }
    wmma_gemm4<<<dim3(CR / 64, NPIX / 128), 256, 0, stream>>>(
        col16, bpack, shbuf, b_rpn, nullptr, nullptr, nullptr, nullptr,
        NPIX, CR, 9 * CF, 1);
    // 7) rpn cls logits (4096x18), 8) rpn deltas (4096x36) -> out directly
    {
        long n = (long)NPIX * 18;
        dense_bias<<<(unsigned)((n + 255) / 256), 256, 0, stream>>>(shbuf, w_cls, b_cls, rpn_cl, NPIX, 18, CR);
        n = (long)NPIX * 36;
        dense_bias<<<(unsigned)((n + 255) / 256), 256, 0, stream>>>(shbuf, w_dlt, b_dlt, out_delta, NPIX, 36, CR);
    }
    // 9) softmax over anchor pairs -> rpn_probs out + fg scores
    rpn_softmax<<<(NA + 255) / 256, 256, 0, stream>>>(rpn_cl, out_probs, scores, NA);
    // 10) exact top-6000
    topk_select<<<(NA + 255) / 256, 256, 0, stream>>>(scores, tv, ti, NA, PRE);
    // 11) anchors + decode + clip + normalize
    decode_props<<<(PRE + 255) / 256, 256, 0, stream>>>(ti, tv, out_delta, props, nsc,
                                                        image_h, image_w, PRE, FW);
    // 12) NMS (1000 picks)
    nms_kernel<<<1, 256, 0, stream>>>(props, nsc, PRE, sel, vld, PC, 0.7f);
    // 13) gather proposals -> out + ws
    gather_props<<<(PC + 255) / 256, 256, 0, stream>>>(props, sel, vld, out_prop, prop_ws, PC);
    // 14) roi_align -> rois f16 (A-matrix for FC1)
    roi_align_kernel<<<dim3(POOL * POOL, PC), CF, 0, stream>>>(xbuf, prop_ws, rois16,
                                                               FH, FW, CF, POOL);
    // 15) FC1 + bn + relu (the 103 GFLOP GEMM)
    {
        long n = (long)KROI * 1024;
        pack_b_f16<<<(unsigned)((n + 255) / 256), 256, 0, stream>>>(w_rc1, bpack, KROI, 1024);
    }
    wmma_gemm4<<<dim3(1024 / 64, (PC + 127) / 128), 256, 0, stream>>>(
        rois16, bpack, h1, b_rc1, g1, be1, m1, v1, PC, 1024, KROI, 3);
    // 16) h1 -> f16
    {
        long n = (long)PC * 1024;
        cvt_f32_f16<<<(unsigned)((n + 255) / 256), 256, 0, stream>>>(h1, h1f16, n);
    }
    // 17) FC2 + bn + relu
    {
        long n = (long)1024 * 1024;
        pack_b_f16<<<(unsigned)((n + 255) / 256), 256, 0, stream>>>(w_rc2, bpack, 1024, 1024);
    }
    wmma_gemm4<<<dim3(1024 / 64, (PC + 127) / 128), 256, 0, stream>>>(
        h1f16, bpack, h2, b_rc2, g2, be2, m2, v2, PC, 1024, 1024, 3);
    // 18) class logits + softmax -> out
    {
        long n = (long)PC * NCLS;
        dense_bias<<<(unsigned)((n + 255) / 256), 256, 0, stream>>>(h2, w_log, b_log, clsl, PC, NCLS, 1024);
        softmax_rows<<<(PC + 255) / 256, 256, 0, stream>>>(clsl, out_cls, PC, NCLS);
        // 19) box deltas -> out
        n = (long)PC * NCLS * 4;
        dense_bias<<<(unsigned)((n + 255) / 256), 256, 0, stream>>>(h2, w_bfc, b_bfc, out_box, PC, NCLS * 4, 1024);
    }
}
